// MultiviewSNF_29429115912454
// MI455X (gfx1250) — compile-verified, matrix operand order checked by reading, not run on hardware
//
#include <hip/hip_runtime.h>

// ---------------------------------------------------------------------------
// MI455X (gfx1250) multiview GCN forward:
//   z_i = spmm(A_i, relu(spmm(A_i, x_i@w1_i)+b1_i) @ w2_i) + b2_i
//   z_f = relu([z0 z1 z2] @ Wf + bf);  q = student-t soft assignment
// Dense GEMMs: V_WMMA_F32_16X16X32_BF16 (fp32 converted RNE->bf16 while
// staging to LDS, f32 accumulate). GEMM2's A operand is already bf16 in
// global memory, so it is staged with GLOBAL_LOAD_ASYNC_TO_LDS_B128
// (ASYNCcnt + s_wait_asynccnt). SpMM is atomic scatter-add into L2-resident
// matrices (h/z fit easily in the 192MB L2). Wave32 everywhere.
// ---------------------------------------------------------------------------

typedef __bf16 v16bf __attribute__((ext_vector_type(16)));
typedef float  v8f   __attribute__((ext_vector_type(8)));

union FragBF { unsigned u[8]; v16bf v; };

// Round-to-nearest-even fp32 -> bf16, packed pair into one dword.
__device__ __forceinline__ unsigned pack_bf16x2(float lo, float hi) {
  unsigned a = __float_as_uint(lo);
  unsigned b = __float_as_uint(hi);
  a += 0x7FFFu + ((a >> 16) & 1u);
  b += 0x7FFFu + ((b >> 16) & 1u);
  return (a >> 16) | (b & 0xFFFF0000u);
}

// LDS uint index j holds K = 2*j ; WMMA 16-bit A/B fragment layout
// (ISA 7.12.2): lanes 0-15 -> K blocks {0..7,16..23}, lanes 16-31 ->
// {8..15,24..31}; per-VGPR pair p maps to uint j below.
__device__ __forceinline__ int frag_j(int p, int half) {
  return ((p >= 4) ? 8 : 0) + half * 4 + (p & 3);
}

// ---- CDNA5 async global->LDS copy (ASYNCcnt) ------------------------------
// Builtin signature (from clang diagnostic): param0 is AS1 (device) pointer
// to a 16-byte vector; param1 is the LDS (AS3) destination.
typedef int v4i __attribute__((vector_size(16)));
typedef __attribute__((address_space(1))) v4i g_v4i;
typedef __attribute__((address_space(3))) v4i l_v4i;

__device__ __forceinline__ void async_g2l_b128(const void* g, void* l) {
#if __has_builtin(__builtin_amdgcn_global_load_async_to_lds_b128)
  __builtin_amdgcn_global_load_async_to_lds_b128(
      (g_v4i*)g, (l_v4i*)l, 0, 0);
#else
  unsigned lds_off = (unsigned)(unsigned long long)l;  // LDS aperture: addr[31:0]
  asm volatile("global_load_async_to_lds_b128 %0, %1, off"
               :: "v"(lds_off), "v"(g) : "memory");
#endif
}

__device__ __forceinline__ void wait_async0() {
#if __has_builtin(__builtin_amdgcn_s_wait_asynccnt)
  __builtin_amdgcn_s_wait_asynccnt(0);
#else
  asm volatile("s_wait_asynccnt 0x0" ::: "memory");
#endif
}

// ---------------- GEMM1: C[N,128] = bf16(A[N,kdim]) @ bf16(B[kdim,128]) ----
__global__ __launch_bounds__(256) void view_gemm1(
    const float* __restrict__ A, const float* __restrict__ B,
    float* __restrict__ C, int nrows, int kdim) {
  __shared__ unsigned lds_a[128][17];   // [row][kpair]
  __shared__ unsigned lds_b[128][17];   // [col][kpair] (transposed)
  const int tid  = threadIdx.x;
  const int lane = tid & 31;
  const int wave = tid >> 5;
  const int ml   = lane & 15;
  const int half = lane >> 4;
  const long mbase = (long)blockIdx.x * 128;

  v8f acc[8];
  for (int t = 0; t < 8; ++t)
    for (int j = 0; j < 8; ++j) acc[t][j] = 0.f;

  for (int kb = 0; kb < kdim; kb += 32) {
    __syncthreads();
#pragma unroll
    for (int it = 0; it < 8; ++it) {          // stage A tile 128x32
      int idx = it * 256 + tid;
      int j = idx & 15, r = idx >> 4;
      long m = mbase + r;
      int k = kb + 2 * j;
      float lo = 0.f, hi = 0.f;
      if (m < nrows) {
        const float* ap = A + m * (long)kdim;
        if (k < kdim)     lo = ap[k];
        if (k + 1 < kdim) hi = ap[k + 1];
      }
      lds_a[r][j] = pack_bf16x2(lo, hi);
    }
#pragma unroll
    for (int it = 0; it < 8; ++it) {          // stage B tile 32x128 (transpose)
      int idx = it * 256 + tid;
      int c = idx & 127, j = idx >> 7;
      int k = kb + 2 * j;
      float lo = (k     < kdim) ? B[(long)k * 128 + c]       : 0.f;
      float hi = (k + 1 < kdim) ? B[(long)(k + 1) * 128 + c] : 0.f;
      lds_b[c][j] = pack_bf16x2(lo, hi);
    }
    __syncthreads();

    FragBF fa;
#pragma unroll
    for (int p = 0; p < 8; ++p) fa.u[p] = lds_a[wave * 16 + ml][frag_j(p, half)];
#pragma unroll
    for (int nt = 0; nt < 8; ++nt) {
      FragBF fb;
#pragma unroll
      for (int p = 0; p < 8; ++p) fb.u[p] = lds_b[nt * 16 + ml][frag_j(p, half)];
      acc[nt] = __builtin_amdgcn_wmma_f32_16x16x32_bf16(
          false, fa.v, false, fb.v, (short)0, acc[nt], false, false);
    }
  }
#pragma unroll
  for (int nt = 0; nt < 8; ++nt) {
    int n = nt * 16 + ml;
#pragma unroll
    for (int r = 0; r < 8; ++r) {
      long m = mbase + wave * 16 + half * 8 + r;
      if (m < nrows) C[m * 128 + n] = acc[nt][r];
    }
  }
}

// ---------------- GEMM2: C[N,64] = Apk(bf16 packed [N,128]) @ bf16(B[128,64])
// A operand staged with CDNA5 async global->LDS B128 copies.
__global__ __launch_bounds__(256) void gemm_h_w2(
    const unsigned* __restrict__ Apk, const float* __restrict__ B,
    float* __restrict__ C, int nrows) {
  __shared__ unsigned lds_a[128][20];   // stride 20 dwords: 16B-aligned rows
  __shared__ unsigned lds_b[64][17];
  const int tid  = threadIdx.x;
  const int lane = tid & 31;
  const int wave = tid >> 5;
  const int ml   = lane & 15;
  const int half = lane >> 4;
  const long mbase = (long)blockIdx.x * 128;

  v8f acc[4];
  for (int t = 0; t < 4; ++t)
    for (int j = 0; j < 8; ++j) acc[t][j] = 0.f;

  for (int kb = 0; kb < 128; kb += 32) {
    __syncthreads();                           // LDS free of prior readers
#pragma unroll
    for (int it = 0; it < 2; ++it) {           // async-stage A tile 128x32 bf16
      int idx = it * 256 + tid;                // 512 x 16B transfers
      int r  = idx >> 2;                       // row 0..127
      int qd = (idx & 3) * 4;                  // dword quad within row
      long m = mbase + r;
      if (m >= nrows) m = nrows - 1;           // clamp: keep EXEC all-ones
      async_g2l_b128(Apk + m * 64 + (kb >> 1) + qd, &lds_a[r][qd]);
    }
#pragma unroll
    for (int it = 0; it < 4; ++it) {           // stage B tile 32x64 (transpose)
      int idx = it * 256 + tid;
      int c = idx & 63, j = idx >> 6;
      int k = kb + 2 * j;
      lds_b[c][j] = pack_bf16x2(B[k * 64 + c], B[(k + 1) * 64 + c]);
    }
    wait_async0();                             // our ASYNCcnt -> 0
    __syncthreads();                           // all waves' transfers visible

    FragBF fa;
#pragma unroll
    for (int p = 0; p < 8; ++p) fa.u[p] = lds_a[wave * 16 + ml][frag_j(p, half)];
#pragma unroll
    for (int nt = 0; nt < 4; ++nt) {
      FragBF fb;
#pragma unroll
      for (int p = 0; p < 8; ++p) fb.u[p] = lds_b[nt * 16 + ml][frag_j(p, half)];
      acc[nt] = __builtin_amdgcn_wmma_f32_16x16x32_bf16(
          false, fa.v, false, fb.v, (short)0, acc[nt], false, false);
    }
  }
#pragma unroll
  for (int nt = 0; nt < 4; ++nt) {
    int n = nt * 16 + ml;
#pragma unroll
    for (int r = 0; r < 8; ++r) {
      long m = mbase + wave * 16 + half * 8 + r;
      if (m < nrows) C[m * 64 + n] = acc[nt][r];
    }
  }
}

// ------------- Fusion: zf[N,64] = relu(concat(z0,z1,z2)[N,192] @ W + bias) --
__global__ __launch_bounds__(256) void fusion_gemm(
    const float* __restrict__ z0, const float* __restrict__ z1,
    const float* __restrict__ z2, const float* __restrict__ W,
    const float* __restrict__ bias, float* __restrict__ C, int nrows) {
  __shared__ unsigned lds_a[128][17];
  __shared__ unsigned lds_b[64][17];
  const int tid  = threadIdx.x;
  const int lane = tid & 31;
  const int wave = tid >> 5;
  const int ml   = lane & 15;
  const int half = lane >> 4;
  const long mbase = (long)blockIdx.x * 128;

  v8f acc[4];
  for (int t = 0; t < 4; ++t)
    for (int j = 0; j < 8; ++j) acc[t][j] = 0.f;

  for (int kb = 0; kb < 192; kb += 32) {
    __syncthreads();
#pragma unroll
    for (int it = 0; it < 8; ++it) {          // stage A from 3 z segments
      int idx = it * 256 + tid;
      int j = idx & 15, r = idx >> 4;
      long m = mbase + r;
      int kc = kb + 2 * j;                    // pair never crosses a segment
      const float* src = (kc < 64) ? z0 : ((kc < 128) ? z1 : z2);
      int cc = kc & 63;
      float lo = 0.f, hi = 0.f;
      if (m < nrows) { lo = src[m * 64 + cc]; hi = src[m * 64 + cc + 1]; }
      lds_a[r][j] = pack_bf16x2(lo, hi);
    }
#pragma unroll
    for (int it = 0; it < 4; ++it) {
      int idx = it * 256 + tid;
      int c = idx & 63, j = idx >> 6;
      int k = kb + 2 * j;
      lds_b[c][j] = pack_bf16x2(W[k * 64 + c], W[(k + 1) * 64 + c]);
    }
    __syncthreads();

    FragBF fa;
#pragma unroll
    for (int p = 0; p < 8; ++p) fa.u[p] = lds_a[wave * 16 + ml][frag_j(p, half)];
#pragma unroll
    for (int nt = 0; nt < 4; ++nt) {
      FragBF fb;
#pragma unroll
      for (int p = 0; p < 8; ++p) fb.u[p] = lds_b[nt * 16 + ml][frag_j(p, half)];
      acc[nt] = __builtin_amdgcn_wmma_f32_16x16x32_bf16(
          false, fa.v, false, fb.v, (short)0, acc[nt], false, false);
    }
  }
#pragma unroll
  for (int nt = 0; nt < 4; ++nt) {
    int n = nt * 16 + ml;
    float bn = bias[n];
#pragma unroll
    for (int r = 0; r < 8; ++r) {
      long m = mbase + wave * 16 + half * 8 + r;
      if (m < nrows) C[m * 64 + n] = fmaxf(acc[nt][r] + bn, 0.f);
    }
  }
}

// ---------------- SpMM scatter (COO): out[row] += val * X[col] -------------
__global__ __launch_bounds__(256) void spmm_scatter128(
    const int* __restrict__ rows, const int* __restrict__ cols,
    const float* __restrict__ vals, const float* __restrict__ X,
    float* __restrict__ out, int nedges) {
  long tid = (long)blockIdx.x * 256 + threadIdx.x;
  long e = tid >> 5;                  // 32 lanes per edge, 4 feats each
  if (e >= nedges) return;
  int fq = (int)(tid & 31);
  int r = rows[e], c = cols[e];
  float v = vals[e];
  float4 xv = ((const float4*)(X + (long)c * 128))[fq];
  float* o = out + (long)r * 128 + fq * 4;
  unsafeAtomicAdd(o + 0, v * xv.x);
  unsafeAtomicAdd(o + 1, v * xv.y);
  unsafeAtomicAdd(o + 2, v * xv.z);
  unsafeAtomicAdd(o + 3, v * xv.w);
}

__global__ __launch_bounds__(256) void spmm_scatter64(
    const int* __restrict__ rows, const int* __restrict__ cols,
    const float* __restrict__ vals, const float* __restrict__ X,
    float* __restrict__ out, int nedges) {
  long tid = (long)blockIdx.x * 256 + threadIdx.x;
  long e = tid >> 4;                  // 16 lanes per edge, 4 feats each
  if (e >= nedges) return;
  int fq = (int)(tid & 15);
  int r = rows[e], c = cols[e];
  float v = vals[e];
  float4 xv = ((const float4*)(X + (long)c * 64))[fq];
  float* o = out + (long)r * 64 + fq * 4;
  unsafeAtomicAdd(o + 0, v * xv.x);
  unsafeAtomicAdd(o + 1, v * xv.y);
  unsafeAtomicAdd(o + 2, v * xv.z);
  unsafeAtomicAdd(o + 3, v * xv.w);
}

// ---------------- helpers ---------------------------------------------------
__global__ __launch_bounds__(256) void fill_bias(
    float* __restrict__ out, const float* __restrict__ b, int mask, long total) {
  long i = (long)blockIdx.x * 256 + threadIdx.x;
  if (i < total) out[i] = b[(int)(i & mask)];
}

__global__ __launch_bounds__(256) void relu_pack_bf16(
    const float* __restrict__ h, unsigned* __restrict__ o, long npairs) {
  long i = (long)blockIdx.x * 256 + threadIdx.x;
  if (i >= npairs) return;
  o[i] = pack_bf16x2(fmaxf(h[2 * i], 0.f), fmaxf(h[2 * i + 1], 0.f));
}

// ---------------- DEC soft assignment --------------------------------------
__global__ __launch_bounds__(256) void soft_assign(
    const float* __restrict__ zf, const float* __restrict__ cl,
    float* __restrict__ q, int n) {
  __shared__ float c[640];            // 10 clusters x 64 dims
  for (int i = threadIdx.x; i < 640; i += 256) c[i] = cl[i];
  __syncthreads();
  int node = blockIdx.x * 256 + threadIdx.x;
  if (node >= n) return;
  float z[64];
  const float4* zp = (const float4*)(zf + (long)node * 64);
#pragma unroll
  for (int j = 0; j < 16; ++j) {
    float4 t = zp[j];
    z[4 * j] = t.x; z[4 * j + 1] = t.y; z[4 * j + 2] = t.z; z[4 * j + 3] = t.w;
  }
  float qq[10], s = 0.f;
#pragma unroll
  for (int k = 0; k < 10; ++k) {
    float d2 = 0.f;
#pragma unroll
    for (int f = 0; f < 64; ++f) {
      float d = z[f] - c[k * 64 + f];
      d2 = fmaf(d, d, d2);
    }
    qq[k] = 1.f / (1.f + d2);
    s += qq[k];
  }
  float inv = 1.f / s;
  float* qo = q + (long)node * 10;
#pragma unroll
  for (int k = 0; k < 10; ++k) qo[k] = qq[k] * inv;
}

// ---------------------------------------------------------------------------
extern "C" void kernel_launch(void* const* d_in, const int* in_sizes, int n_in,
                              void* d_out, int out_size, void* d_ws, size_t ws_size,
                              hipStream_t stream) {
  const int HID = 128, EMB = 64;
  const int din0 = in_sizes[4] / HID;         // w1_0 = [din0,128]
  const int N = in_sizes[0] / din0;           // x0 = [N,din0]

  float* out = (float*)d_out;
  float* zseg[3] = { out, out + (size_t)N * EMB, out + (size_t)2 * N * EMB };
  float* zfused = out + (size_t)3 * N * EMB;
  float* qout   = out + (size_t)4 * N * EMB;

  // workspace: xw[N,128] f32 | hpre[N,128] f32 | hbf[N,128] bf16 (N*64 dwords)
  // hw2[N,64] aliases xw (dead by then). Total = N*320*4 B = 64 MB.
  float*    xw   = (float*)d_ws;
  float*    hpre = xw + (size_t)N * HID;
  unsigned* hbf  = (unsigned*)(hpre + (size_t)N * HID);
  float*    hw2  = xw;

  const dim3 blk(256);
  const int gemmBlocks = (N + 127) / 128;

  for (int i = 0; i < 3; ++i) {
    const float* x    = (const float*)d_in[8 * i + 0];
    const int*   rows = (const int*)  d_in[8 * i + 1];
    const int*   cols = (const int*)  d_in[8 * i + 2];
    const float* vals = (const float*)d_in[8 * i + 3];
    const float* w1   = (const float*)d_in[8 * i + 4];
    const float* b1   = (const float*)d_in[8 * i + 5];
    const float* w2   = (const float*)d_in[8 * i + 6];
    const float* b2   = (const float*)d_in[8 * i + 7];
    const int E   = in_sizes[8 * i + 1];
    const int din = in_sizes[8 * i + 4] / HID;

    view_gemm1<<<gemmBlocks, blk, 0, stream>>>(x, w1, xw, N, din);

    long tot1 = (long)N * HID;
    fill_bias<<<(int)((tot1 + 255) / 256), blk, 0, stream>>>(hpre, b1, HID - 1, tot1);
    long sc1 = (long)E * 32;
    spmm_scatter128<<<(int)((sc1 + 255) / 256), blk, 0, stream>>>(
        rows, cols, vals, xw, hpre, E);

    long np = (long)N * (HID / 2);
    relu_pack_bf16<<<(int)((np + 255) / 256), blk, 0, stream>>>(hpre, hbf, np);

    gemm_h_w2<<<gemmBlocks, blk, 0, stream>>>(hbf, w2, hw2, N);

    long tot2 = (long)N * EMB;
    fill_bias<<<(int)((tot2 + 255) / 256), blk, 0, stream>>>(zseg[i], b2, EMB - 1, tot2);
    long sc2 = (long)E * 16;
    spmm_scatter64<<<(int)((sc2 + 255) / 256), blk, 0, stream>>>(
        rows, cols, vals, hw2, zseg[i], E);
  }

  fusion_gemm<<<gemmBlocks, blk, 0, stream>>>(
      zseg[0], zseg[1], zseg[2],
      (const float*)d_in[24], (const float*)d_in[25], zfused, N);

  soft_assign<<<(N + 255) / 256, blk, 0, stream>>>(
      zfused, (const float*)d_in[26], qout, N);
}
